// GPTModel_27298812133609
// MI455X (gfx1250) — matrix operand order repaired and checked by reading, not compiled, not run on hardware
//
#include <hip/hip_runtime.h>

// ---------------------------------------------------------------------------
// GPT forward for MI455X (gfx1250, wave32, v_wmma_f32_16x16x32_bf16).
// Weights converted f32->bf16 AND pre-transposed (B^T, N-major) once per
// launch, so every WMMA B-fragment is two contiguous global_load_b128 -- no
// LDS staging, no barriers in the GEMM. bf16 weights stay L2-resident
// (192 MB). QKV fused into one GEMM (Q pre-scaled by 1/sqrt(HD)); V written
// directly transposed for attention. Flash attention, 4 query tiles (same
// head) per 128-thread block; causal mask only on the diagonal key-block.
// CDNA5 async global->LDS staging of Q tiles (guarded by __has_builtin).
// ---------------------------------------------------------------------------

#define DEVFN __device__ __forceinline__

typedef __attribute__((ext_vector_type(16))) __bf16 v16bf;
typedef __attribute__((ext_vector_type(8)))  __bf16 v8bf;
typedef __attribute__((ext_vector_type(8)))  float  v8f;
typedef __attribute__((ext_vector_type(4)))  int    v4i;

#if defined(__has_builtin)
#if __has_builtin(__builtin_amdgcn_global_load_async_to_lds_b128) && \
    __has_builtin(__builtin_amdgcn_s_wait_asynccnt)
#define USE_ASYNC_LDS 1
#endif
#endif

DEVFN unsigned short f32_to_bf16_raw(float f) {
  unsigned int u = __float_as_uint(f);
  u += 0x7FFFu + ((u >> 16) & 1u);       // round-to-nearest-even
  return (unsigned short)(u >> 16);
}

DEVFN v16bf mk16(v8bf lo, v8bf hi) {
  return __builtin_shufflevector(lo, hi, 0, 1, 2, 3, 4, 5, 6, 7,
                                 8, 9, 10, 11, 12, 13, 14, 15);
}

DEVFN v8bf ld8(const unsigned short* p) { return *(const v8bf*)p; }

DEVFN v8f wmma_bf16(v16bf a, v16bf b, v8f c) {
  return __builtin_amdgcn_wmma_f32_16x16x32_bf16(false, a, false, b,
                                                 (short)0, c, false, false);
}

DEVFN float gelu_f(float x) {
  return 0.5f * x * (1.0f + tanhf(0.7978845608028654f * (x + 0.044715f * x * x * x)));
}

// ---------------------------------------------------------------------------
// Transpose + f32->bf16 weight conversion: dst[n*K + k] = bf16(src[k*N + n])
// 64x64 tiles via LDS. blockIdx.z = layer. Rows n in [N, Npad) zero-filled.
// ---------------------------------------------------------------------------
__global__ __launch_bounds__(256)
void transpose_cvt_kernel(const float* __restrict__ src0,
                          unsigned short* __restrict__ dst0,
                          int K, int N, int Npad) {
  __shared__ unsigned short tls[64 * 72];
  const float* src = src0 + (size_t)blockIdx.z * K * N;
  unsigned short* dst = dst0 + (size_t)blockIdx.z * Npad * K;
  const int n0 = blockIdx.x * 64, k0 = blockIdx.y * 64;
#pragma unroll
  for (int i = 0; i < 16; ++i) {
    int e = i * 256 + threadIdx.x;
    int kk = e >> 6, nn = e & 63;
    int gn = n0 + nn;
    float v = (gn < N) ? src[(size_t)(k0 + kk) * N + gn] : 0.f;
    tls[nn * 72 + kk] = f32_to_bf16_raw(v);
  }
  __syncthreads();
#pragma unroll
  for (int i = 0; i < 16; ++i) {
    int e = i * 256 + threadIdx.x;
    int nn = e >> 6, kk = e & 63;
    dst[(size_t)(n0 + nn) * K + k0 + kk] = tls[nn * 72 + kk];
  }
}

// Fused QKV weight: logical B[k][n], n in [0,2304): col n of wq/wk/wv.
// dst layout: [l][n][k], k-major (B^T).
__global__ __launch_bounds__(256)
void qkv_transpose_cvt_kernel(const float* __restrict__ wq,
                              const float* __restrict__ wk,
                              const float* __restrict__ wv,
                              unsigned short* __restrict__ dst0, int E) {
  __shared__ unsigned short tls[64 * 72];
  const int l = blockIdx.z;
  unsigned short* dst = dst0 + (size_t)l * 3 * E * E;
  const int n0 = blockIdx.x * 64, k0 = blockIdx.y * 64;
#pragma unroll
  for (int i = 0; i < 16; ++i) {
    int e = i * 256 + threadIdx.x;
    int kk = e >> 6, nn = e & 63;
    int gn = n0 + nn;
    int sel = gn / E;                 // 0=q,1=k,2=v (tiles never straddle: 768%64==0)
    int col = gn - sel * E;
    const float* w = (sel == 0) ? wq : (sel == 1) ? wk : wv;
    tls[nn * 72 + kk] = f32_to_bf16_raw(w[(size_t)l * E * E + (size_t)(k0 + kk) * E + col]);
  }
  __syncthreads();
#pragma unroll
  for (int i = 0; i < 16; ++i) {
    int e = i * 256 + threadIdx.x;
    int nn = e >> 6, kk = e & 63;
    dst[(size_t)(n0 + nn) * E + k0 + kk] = tls[nn * 72 + kk];
  }
}

// ---------------------------------------------------------------------------
// Embedding: h[row, e] = tok_emb[x[row], e] + sinusoidal(pos, e)
// ---------------------------------------------------------------------------
__global__ void embed_kernel(const int* __restrict__ x,
                             const float* __restrict__ tok,
                             float* __restrict__ h, int n, int e, int total) {
  int idx = blockIdx.x * 256 + threadIdx.x;
  if (idx >= total) return;
  int col = idx % e;
  int row = idx / e;
  int pos = row % n;
  int t = x[row];
  float ang = (float)pos * __powf(10000.0f, -2.0f * (float)col / (float)e);
  float pe = (col & 1) ? __cosf(ang) : __sinf(ang);
  h[idx] = tok[(size_t)t * e + col] + pe;
}

// ---------------------------------------------------------------------------
// LayerNorm (row per block), bf16 output
// ---------------------------------------------------------------------------
__global__ __launch_bounds__(256)
void layernorm_kernel(const float* __restrict__ X, const float* __restrict__ gw,
                      const float* __restrict__ bw, unsigned short* __restrict__ Y,
                      int ncols) {
  __shared__ float red[32];
  int row = blockIdx.x;
  const float* x = X + (size_t)row * ncols;
  float s = 0.f, s2 = 0.f;
  for (int c = threadIdx.x; c < ncols; c += 256) { float v = x[c]; s += v; s2 += v * v; }
#pragma unroll
  for (int off = 1; off < 32; off <<= 1) {
    s += __shfl_xor(s, off, 32);
    s2 += __shfl_xor(s2, off, 32);
  }
  int wv = threadIdx.x >> 5, lane = threadIdx.x & 31;
  if (lane == 0) { red[wv] = s; red[8 + wv] = s2; }
  __syncthreads();
  if (threadIdx.x == 0) {
    float a = 0.f, b = 0.f;
#pragma unroll
    for (int i = 0; i < 8; ++i) { a += red[i]; b += red[8 + i]; }
    red[16] = a; red[17] = b;
  }
  __syncthreads();
  float mean = red[16] / (float)ncols;
  float var = red[17] / (float)ncols - mean * mean;
  float inv = rsqrtf(var + 1e-5f);
  for (int c = threadIdx.x; c < ncols; c += 256) {
    float v = (x[c] - mean) * inv * gw[c] + bw[c];
    Y[(size_t)row * ncols + c] = f32_to_bf16_raw(v);
  }
}

// ---------------------------------------------------------------------------
// WMMA GEMM, no LDS: C(MxN) = A(MxK bf16 row-major) * B (given as B^T, NxK).
// 256 threads = 8 waves; wave tile 16(M) x 64(N); block tile 128 x 64.
// K-step 64 -> 8 WMMAs/iter/wave. A and B fragments are contiguous
// global_load_b128 pairs (B^T rows are K-major).
// MODE 1: bias+GELU -> bf16.  MODE 2: bias, f32 residual +=.
// MODE 3: f32 store, N-bounds.  MODE 4: fused QKV split (q scaled by 1/8 and
// stored row-major with k; v stored transposed [b,h,hd][pos] for attention).
// ---------------------------------------------------------------------------
template <int MODE>
__global__ __launch_bounds__(256)
void gemm_kernel(const unsigned short* __restrict__ A,
                 const unsigned short* __restrict__ Bt,
                 const float* __restrict__ bias,
                 void* __restrict__ out,
                 unsigned short* __restrict__ out2,
                 float* __restrict__ resid,
                 int M, int K, int N, int nTok) {
  const int lane = threadIdx.x & 31;
  const int wave = threadIdx.x >> 5;
  const int n0 = blockIdx.x * 64;
  const int m0 = blockIdx.y * 128 + wave * 16;
  const int g = lane >> 4, ln = lane & 15, kb8 = g * 8;

  v8f acc[4];
#pragma unroll
  for (int t = 0; t < 4; ++t)
#pragma unroll
    for (int r = 0; r < 8; ++r) acc[t][r] = 0.f;

  const unsigned short* arow = A + (size_t)(m0 + ln) * K;
  const unsigned short* brow[4];
#pragma unroll
  for (int t = 0; t < 4; ++t)
    brow[t] = Bt + (size_t)(n0 + t * 16 + ln) * K;

  for (int k0 = 0; k0 < K; k0 += 64) {
    if (k0 + 64 < K) {
      __builtin_prefetch(arow + k0 + 64, 0, 1);
      __builtin_prefetch(brow[0] + k0 + 64, 0, 1);
    }
    v16bf a0 = mk16(ld8(arow + k0 + g * 16), ld8(arow + k0 + g * 16 + 8));
    v16bf a1 = mk16(ld8(arow + k0 + 32 + g * 16), ld8(arow + k0 + 32 + g * 16 + 8));
#pragma unroll
    for (int t = 0; t < 4; ++t) {
      const unsigned short* bp = brow[t] + k0 + g * 16;
      v16bf b0 = mk16(ld8(bp), ld8(bp + 8));
      acc[t] = wmma_bf16(a0, b0, acc[t]);
      v16bf b1 = mk16(ld8(bp + 32), ld8(bp + 40));
      acc[t] = wmma_bf16(a1, b1, acc[t]);
    }
  }

  // C layout: vgpr r, lane l -> row = r + (l>>4)*8, col = l&15
#pragma unroll
  for (int t = 0; t < 4; ++t) {
    int n = n0 + t * 16 + ln;
    if (n >= N) continue;
    float bv = (MODE == 1 || MODE == 2) ? bias[n] : 0.f;
#pragma unroll
    for (int r = 0; r < 8; ++r) {
      int m = m0 + g * 8 + r;
      float v = acc[t][r] + bv;
      if (MODE == 1) {
        ((unsigned short*)out)[(size_t)m * N + n] = f32_to_bf16_raw(gelu_f(v));
      } else if (MODE == 2) {
        resid[(size_t)m * N + n] += v;
      } else if (MODE == 3) {
        ((float*)out)[(size_t)m * N + n] = v;
      } else {  // MODE 4: fused QKV
        if (n < 1536) {
          float q = (n < 768) ? v * 0.125f : v;   // fold 1/sqrt(64) into Q
          ((unsigned short*)out)[(size_t)m * 1536 + n] = f32_to_bf16_raw(q);
        } else {
          int hd = n - 1536;              // h*64 + hd, 0..767
          int bb = m / nTok, pos = m - bb * nTok;
          out2[((size_t)bb * 768 + hd) * nTok + pos] = f32_to_bf16_raw(v);
        }
      }
    }
  }
}

// ---------------------------------------------------------------------------
// Flash attention: 4 waves/block = 4 adjacent 16-query tiles of SAME (b,h)
// (shared K/V stream in L0). QK: [M][1536] row-major (q pre-scaled at col 0,
// k at 768). Vt: [b*H*64 + h*64 + hd][pos]. Causal mask only applied on the
// single diagonal key-block; sub-diagonal blocks are mask-free.
// ---------------------------------------------------------------------------
__global__ __launch_bounds__(128)
void attn_kernel(const unsigned short* __restrict__ QK,
                 const unsigned short* __restrict__ Vt,
                 unsigned short* __restrict__ Op,
                 int nTok, int Hh) {
  __shared__ unsigned short PtS[4][16 * 32];
#if USE_ASYNC_LDS
  __shared__ unsigned short QtS[4][16 * 64];
#endif
  const int lane = threadIdx.x & 31;
  const int wave = threadIdx.x >> 5;
  const int qt = blockIdx.x * 4 + wave;
  const int hh = blockIdx.y;
  const int bb = blockIdx.z;
  const size_t rowbase = (size_t)bb * nTok;
  const int g = lane >> 4, ln15 = lane & 15, kb8 = g * 8;
  const int ldqk = 1536;

  v16bf aq0, aq1;
#if USE_ASYNC_LDS
  {
    // CDNA5 async global->LDS staging of the 16x64 Q tile (ASYNCcnt-tracked)
    const unsigned short* qg = QK + (rowbase + qt * 16) * ldqk + hh * 64;
#pragma unroll
    for (int c = 0; c < 4; ++c) {
      int chunk = c * 32 + lane;          // 0..127 chunks of 16B
      int row = chunk >> 3, col = (chunk & 7) * 8;
      __builtin_amdgcn_global_load_async_to_lds_b128(
          (__attribute__((address_space(1))) v4i*)(uintptr_t)(
              qg + (size_t)row * ldqk + col),
          (__attribute__((address_space(3))) v4i*)(unsigned int)(uintptr_t)(
              &QtS[wave][row * 64 + col]),
          0, 0);
    }
    __builtin_amdgcn_s_wait_asynccnt(0);
    const unsigned short* qrow = &QtS[wave][ln15 * 64];
    aq0 = mk16(ld8(qrow + kb8), ld8(qrow + 16 + kb8));
    aq1 = mk16(ld8(qrow + 32 + kb8), ld8(qrow + 48 + kb8));
  }
#else
  {
    const unsigned short* qrow = QK + (rowbase + qt * 16 + ln15) * ldqk + hh * 64;
    aq0 = mk16(ld8(qrow + kb8), ld8(qrow + 16 + kb8));
    aq1 = mk16(ld8(qrow + 32 + kb8), ld8(qrow + 48 + kb8));
  }
#endif

  v8f o_acc[4];
  float m_r[8], l_r[8];
#pragma unroll
  for (int t = 0; t < 4; ++t)
#pragma unroll
    for (int r = 0; r < 8; ++r) o_acc[t][r] = 0.f;
#pragma unroll
  for (int r = 0; r < 8; ++r) { m_r[r] = -1e30f; l_r[r] = 0.f; }

  // hoisted per-lane row pointers: K (score B-fragments) and V (PV B-fragments)
  const int k0f = g * 16;
  const unsigned short* kr0 = QK + (rowbase + ln15) * ldqk + 768 + hh * 64;
  const unsigned short* kr1 = kr0 + (size_t)16 * ldqk;
  const unsigned short* vbase = Vt + ((size_t)bb * Hh + hh) * 64 * nTok;
  const unsigned short* vb[4];
#pragma unroll
  for (int t2 = 0; t2 < 4; ++t2)
    vb[t2] = vbase + (size_t)(t2 * 16 + ln15) * nTok + k0f;

  auto process_block = [&](int kpos0, bool masked) {
    // scores: two 16x16 tiles; B = K^T fragments straight from global
    v8f s[2];
    const unsigned short* kt[2] = { kr0 + (size_t)kpos0 * ldqk,
                                    kr1 + (size_t)kpos0 * ldqk };
#pragma unroll
    for (int t = 0; t < 2; ++t) {
#pragma unroll
      for (int r = 0; r < 8; ++r) s[t][r] = 0.f;
      s[t] = wmma_bf16(aq0, mk16(ld8(kt[t] + kb8), ld8(kt[t] + 16 + kb8)), s[t]);
      s[t] = wmma_bf16(aq1, mk16(ld8(kt[t] + 32 + kb8), ld8(kt[t] + 48 + kb8)), s[t]);
    }

    // online softmax (scale already folded into Q)
#pragma unroll
    for (int r = 0; r < 8; ++r) {
      if (masked) {
        int qpos = qt * 16 + g * 8 + r;
#pragma unroll
        for (int t = 0; t < 2; ++t) {
          int kpos = kpos0 + t * 16 + ln15;
          if (kpos > qpos) s[t][r] = -1e30f;
        }
      }
      float rv = fmaxf(s[0][r], s[1][r]);
#pragma unroll
      for (int off = 1; off < 16; off <<= 1) rv = fmaxf(rv, __shfl_xor(rv, off, 16));
      float mnew = fmaxf(m_r[r], rv);
      float p0 = __expf(s[0][r] - mnew);
      float p1 = __expf(s[1][r] - mnew);
      s[0][r] = p0; s[1][r] = p1;
      float rs = p0 + p1;
#pragma unroll
      for (int off = 1; off < 16; off <<= 1) rs += __shfl_xor(rs, off, 16);
      float corr = __expf(m_r[r] - mnew);
      l_r[r] = l_r[r] * corr + rs;
      m_r[r] = mnew;
#pragma unroll
      for (int t2 = 0; t2 < 4; ++t2) o_acc[t2][r] *= corr;
    }

    // restage P (C-layout -> A-layout) through LDS, then P*V
#pragma unroll
    for (int r = 0; r < 8; ++r) {
      PtS[wave][(g * 8 + r) * 32 + ln15]      = f32_to_bf16_raw(s[0][r]);
      PtS[wave][(g * 8 + r) * 32 + 16 + ln15] = f32_to_bf16_raw(s[1][r]);
    }
    const unsigned short* pp = &PtS[wave][ln15 * 32 + k0f];
    v16bf ap = mk16(ld8(pp), ld8(pp + 8));
#pragma unroll
    for (int t2 = 0; t2 < 4; ++t2) {
      const unsigned short* vp = vb[t2] + kpos0;
      o_acc[t2] = wmma_bf16(ap, mk16(ld8(vp), ld8(vp + 8)), o_acc[t2]);
    }
  };

  const int kb_full = qt >> 1;                 // sub-diagonal 32-key blocks
  for (int kb = 0; kb < kb_full; ++kb) process_block(kb * 32, false);
  process_block(kb_full * 32, true);           // diagonal block (masked)

#pragma unroll
  for (int t2 = 0; t2 < 4; ++t2) {
#pragma unroll
    for (int r = 0; r < 8; ++r) {
      int mrow = qt * 16 + g * 8 + r;
      float ov = o_acc[t2][r] / l_r[r];
      Op[(rowbase + mrow) * 768 + hh * 64 + t2 * 16 + ln15] = f32_to_bf16_raw(ov);
    }
  }
}

// ---------------------------------------------------------------------------
// Host orchestration
// ---------------------------------------------------------------------------
extern "C" void kernel_launch(void* const* d_in, const int* in_sizes, int n_in,
                              void* d_out, int out_size, void* d_ws, size_t ws_size,
                              hipStream_t stream) {
  (void)in_sizes; (void)n_in; (void)out_size; (void)ws_size;
  const int Vv = 50257, Ee = 768, Nn = 1024, Hh = 12, Ll = 6, Ff = 3072, Bb = 2;
  const int M = Bb * Nn;            // 2048
  const int Vpad = 50304;           // 786*64

  const int*   x     = (const int*)d_in[0];
  const float* tok   = (const float*)d_in[1];
  const float* wq    = (const float*)d_in[2];
  const float* wk    = (const float*)d_in[3];
  const float* wv    = (const float*)d_in[4];
  const float* wo    = (const float*)d_in[5];
  const float* bo    = (const float*)d_in[6];
  const float* ln1g  = (const float*)d_in[7];
  const float* ln1b  = (const float*)d_in[8];
  const float* ln2g  = (const float*)d_in[9];
  const float* ln2b  = (const float*)d_in[10];
  const float* w1    = (const float*)d_in[11];
  const float* b1    = (const float*)d_in[12];
  const float* w2    = (const float*)d_in[13];
  const float* b2    = (const float*)d_in[14];
  const float* lnfg  = (const float*)d_in[15];
  const float* lnfb  = (const float*)d_in[16];
  const float* whead = (const float*)d_in[17];

  char* base = (char*)d_ws;
  size_t off = 0;
  auto carve = [&](size_t bytes) -> void* {
    void* p = base + off;
    off += (bytes + 255) & ~(size_t)255;
    return p;
  };
  unsigned short* wqkvT = (unsigned short*)carve((size_t)Ll * 3 * Ee * Ee * 2);
  unsigned short* woT   = (unsigned short*)carve((size_t)Ll * Ee * Ee * 2);
  unsigned short* w1T   = (unsigned short*)carve((size_t)Ll * Ee * Ff * 2);
  unsigned short* w2T   = (unsigned short*)carve((size_t)Ll * Ff * Ee * 2);
  unsigned short* whT   = (unsigned short*)carve((size_t)Vpad * Ee * 2);
  float*          h     = (float*)carve((size_t)M * Ee * 4);
  unsigned short* yB    = (unsigned short*)carve((size_t)M * Ee * 2);
  unsigned short* qkB   = (unsigned short*)carve((size_t)M * 1536 * 2);
  unsigned short* vT    = (unsigned short*)carve((size_t)Bb * Hh * 64 * Nn * 2);
  unsigned short* aoB   = (unsigned short*)carve((size_t)M * Ee * 2);
  unsigned short* f1B   = (unsigned short*)carve((size_t)M * Ff * 2);

  // weight conversion + transpose (bf16, B^T layout)
  qkv_transpose_cvt_kernel<<<dim3(36, 12, Ll), 256, 0, stream>>>(wq, wk, wv, wqkvT, Ee);
  transpose_cvt_kernel<<<dim3(12, 12, Ll), 256, 0, stream>>>(wo, woT, Ee, Ee, Ee);
  transpose_cvt_kernel<<<dim3(48, 12, Ll), 256, 0, stream>>>(w1, w1T, Ee, Ff, Ff);
  transpose_cvt_kernel<<<dim3(12, 48, Ll), 256, 0, stream>>>(w2, w2T, Ff, Ee, Ee);
  transpose_cvt_kernel<<<dim3(Vpad / 64, 12, 1), 256, 0, stream>>>(whead, whT, Ee, Vv, Vpad);

  int total = M * Ee;
  embed_kernel<<<dim3((total + 255) / 256), 256, 0, stream>>>(x, tok, h, Nn, Ee, total);

  dim3 gQKV(36, M / 128);            // N=2304
  dim3 gE(12, M / 128);              // N=768
  dim3 gF(48, M / 128);              // N=3072
  dim3 gA(Nn / 64, Hh, Bb);          // 16 x 12 x 2

  for (int l = 0; l < Ll; ++l) {
    layernorm_kernel<<<M, 256, 0, stream>>>(h, ln1g + (size_t)l * Ee, ln1b + (size_t)l * Ee, yB, Ee);
    gemm_kernel<4><<<gQKV, 256, 0, stream>>>(yB, wqkvT + (size_t)l * 3 * Ee * Ee, nullptr,
                                             qkB, vT, nullptr, M, Ee, 3 * Ee, Nn);
    attn_kernel<<<gA, 128, 0, stream>>>(qkB, vT, aoB, Nn, Hh);
    gemm_kernel<2><<<gE, 256, 0, stream>>>(aoB, woT + (size_t)l * Ee * Ee, bo + (size_t)l * Ee,
                                           nullptr, nullptr, h, M, Ee, Ee, Nn);
    layernorm_kernel<<<M, 256, 0, stream>>>(h, ln2g + (size_t)l * Ee, ln2b + (size_t)l * Ee, yB, Ee);
    gemm_kernel<1><<<gF, 256, 0, stream>>>(yB, w1T + (size_t)l * Ee * Ff, b1 + (size_t)l * Ff,
                                           f1B, nullptr, nullptr, M, Ee, Ff, Nn);
    gemm_kernel<2><<<gE, 256, 0, stream>>>(f1B, w2T + (size_t)l * Ff * Ee, b2 + (size_t)l * Ee,
                                           nullptr, nullptr, h, M, Ff, Ee, Nn);
  }

  layernorm_kernel<<<M, 256, 0, stream>>>(h, lnfg, lnfb, yB, Ee);
  dim3 gV(Vpad / 64, M / 128);       // 786 x 16
  gemm_kernel<3><<<gV, 256, 0, stream>>>(yB, whT, nullptr, (float*)d_out, nullptr, nullptr,
                                         M, Ee, Vv, Nn);
}
